// SelfAttention_59021440582408
// MI455X (gfx1250) — matrix operand order repaired and checked by reading, not compiled, not run on hardware
//
#include <hip/hip_runtime.h>
#include <hip/hip_bf16.h>

// ---------------------------------------------------------------------------
// Fused self-attention for MI455X (gfx1250, wave32, WMMA f16 16x16x32).
//   DH = D@H  (f32 -> f16 "DHT"[t][64]);  lb[x] = -0.5*||DH[:,x]||^2
//   pass1: c[t] = rowmax(S+lb) + ln(rowsum exp(...))     (S via WMMA, never stored)
//   pass2: Z[:,xb..] = l2 * Hh @ exp(S + lb - c[t])      (flash-style fused, LDS dbl-buf)
// ---------------------------------------------------------------------------

typedef _Float16 h8v  __attribute__((ext_vector_type(8)));
typedef _Float16 v16h __attribute__((ext_vector_type(16)));
typedef float    v8f  __attribute__((ext_vector_type(8)));

union U16 { v16h v; h8v h[2]; };

#define T_DIM 8192
#define D_DIM 256
#define N_DIM 64

static __device__ __forceinline__ v8f wmma_f16(v16h a, v16h b, v8f c) {
  // D = A*B + C, f32 accum
  return __builtin_amdgcn_wmma_f32_16x16x32_f16(false, a, false, b, (short)0, c, false, false);
}

// ---------------- prep: DH = D @ H ; DHT (f16) ; DHf32 ----------------------
__global__ void __launch_bounds__(256)
k_dh(const float* __restrict__ D, const float* __restrict__ H,
     float* __restrict__ DHf, _Float16* __restrict__ DHT) {
  const int t = blockIdx.x * 64 + threadIdx.x;  // coalesced along T
  const int n = blockIdx.y * 4 + threadIdx.y;
  float acc = 0.f;
#pragma unroll 4
  for (int k = 0; k < D_DIM; ++k)
    acc = fmaf(D[n * D_DIM + k], H[k * T_DIM + t], acc);
  DHf[n * T_DIM + t] = acc;
  DHT[t * N_DIM + n] = (_Float16)acc;
}

// ---------------- lb[x] = -0.5 * sum_n DH[n][x]^2 ---------------------------
__global__ void __launch_bounds__(256)
k_lb(const float* __restrict__ DHf, float* __restrict__ lb) {
  const int t = blockIdx.x * blockDim.x + threadIdx.x;
  float s = 0.f;
#pragma unroll 8
  for (int n = 0; n < N_DIM; ++n) { float v = DHf[n * T_DIM + t]; s = fmaf(v, v, s); }
  lb[t] = -0.5f * s;
}

// ---------------- Hh = f16(H) ----------------------------------------------
__global__ void __launch_bounds__(256)
k_hh(const float* __restrict__ H, _Float16* __restrict__ Hh) {
  const int i = blockIdx.x * blockDim.x + threadIdx.x;
  Hh[i] = (_Float16)H[i];
}

// ---------------- pass 1: per-row softmax stats c[t] = m + ln(sum) ----------
__global__ void __launch_bounds__(256)
k_rowstats(const _Float16* __restrict__ DHT, const float* __restrict__ lb,
           float* __restrict__ cadj) {
  const int lane = threadIdx.x & 31;
  const int w    = threadIdx.x >> 5;
  const int hi   = lane >> 4;
  const int m16  = lane & 15;
  const int tbase = blockIdx.x * 128 + 16 * w;   // this wave's 16 query rows

  // A tile (fixed for whole kernel): M=t (16), K=n (64) -> two v16h
  const _Float16* ap = DHT + (tbase + m16) * N_DIM + hi * 16;
  U16 a0, a1;
  a0.h[0] = *(const h8v*)(ap);        a0.h[1] = *(const h8v*)(ap + 8);
  a1.h[0] = *(const h8v*)(ap + 32);   a1.h[1] = *(const h8v*)(ap + 40);

  float mx[8], sm[8];
#pragma unroll
  for (int r = 0; r < 8; ++r) { mx[r] = -3.0e38f; sm[r] = 0.f; }

  // pass a: row max of (S + lb)
  for (int xb = 0; xb < T_DIM; xb += 16) {
    const _Float16* bp = DHT + (xb + m16) * N_DIM + hi * 16;
    U16 b0, b1;
    b0.h[0] = *(const h8v*)(bp);       b0.h[1] = *(const h8v*)(bp + 8);
    b1.h[0] = *(const h8v*)(bp + 32);  b1.h[1] = *(const h8v*)(bp + 40);
    v8f s = {};
    s = wmma_f16(a0.v, b0.v, s);
    s = wmma_f16(a1.v, b1.v, s);
    const float lbv = lb[xb + m16];
#pragma unroll
    for (int r = 0; r < 8; ++r) mx[r] = fmaxf(mx[r], s[r] + lbv);
  }
#pragma unroll
  for (int r = 0; r < 8; ++r)
    for (int off = 8; off >= 1; off >>= 1)
      mx[r] = fmaxf(mx[r], __shfl_xor(mx[r], off, 32));  // reduce within 16-lane halves

  // pass b: row sum of exp(S + lb - m)
  for (int xb = 0; xb < T_DIM; xb += 16) {
    const _Float16* bp = DHT + (xb + m16) * N_DIM + hi * 16;
    U16 b0, b1;
    b0.h[0] = *(const h8v*)(bp);       b0.h[1] = *(const h8v*)(bp + 8);
    b1.h[0] = *(const h8v*)(bp + 32);  b1.h[1] = *(const h8v*)(bp + 40);
    v8f s = {};
    s = wmma_f16(a0.v, b0.v, s);
    s = wmma_f16(a1.v, b1.v, s);
    const float lbv = lb[xb + m16];
#pragma unroll
    for (int r = 0; r < 8; ++r) sm[r] += __expf(s[r] + lbv - mx[r]);
  }
#pragma unroll
  for (int r = 0; r < 8; ++r)
    for (int off = 8; off >= 1; off >>= 1)
      sm[r] += __shfl_xor(sm[r], off, 32);

  if (m16 == 0) {
#pragma unroll
    for (int r = 0; r < 8; ++r)
      cadj[tbase + r + hi * 8] = mx[r] + __logf(sm[r]);
  }
}

// ---------------- pass 2: Z = l2 * Hh @ exp(S + lb - c) ---------------------
// Block: 256 thr (8 waves), owns x-cols [xb, xb+32), all 256 c-rows.
// Per 64-t chunk: phase A builds P' tile (64t x 32x, f16) in LDS via WMMA+exp,
// phase B accumulates Z with WMMA. Double-buffered, one barrier per chunk.
__global__ void __launch_bounds__(256)
k_attn(const _Float16* __restrict__ DHT, const _Float16* __restrict__ Hh,
       const float* __restrict__ lb, const float* __restrict__ cadj,
       const float* __restrict__ l2, float* __restrict__ Z) {
  __shared__ __align__(16) _Float16 Pt[2][32][72];   // [buf][x_local][t_local], pitch 72 halves

  const int lane = threadIdx.x & 31;
  const int w    = threadIdx.x >> 5;
  const int hi   = lane >> 4;
  const int m16  = lane & 15;
  const int xb   = blockIdx.x * 32;

  // phase-A tile assignment: 8 waves cover 4 t-subtiles x 2 x-subtiles of the 64x32 chunk
  const int ttl = (w >> 1) * 16;
  const int xl  = (w & 1) * 16;

  // S-phase B operand (DH for this block's x columns) is loop invariant
  U16 sb0, sb1;
  {
    const _Float16* bp = DHT + (xb + xl + m16) * N_DIM + hi * 16;
    sb0.h[0] = *(const h8v*)(bp);       sb0.h[1] = *(const h8v*)(bp + 8);
    sb1.h[0] = *(const h8v*)(bp + 32);  sb1.h[1] = *(const h8v*)(bp + 40);
  }
  const float lbv = lb[xb + xl + m16];

  // Z accumulators: wave w owns c rows [32w, 32w+32) x x cols [xb, xb+32)
  v8f acc[2][2];
#pragma unroll
  for (int i = 0; i < 2; ++i)
#pragma unroll
    for (int j = 0; j < 2; ++j) acc[i][j] = {};

  auto phaseA = [&](int t0, int buf) {
    const _Float16* ap = DHT + (t0 + ttl + m16) * N_DIM + hi * 16;
    U16 a0, a1;
    a0.h[0] = *(const h8v*)(ap);        a0.h[1] = *(const h8v*)(ap + 8);
    a1.h[0] = *(const h8v*)(ap + 32);   a1.h[1] = *(const h8v*)(ap + 40);
    v8f s = {};
    s = wmma_f16(a0.v, sb0.v, s);
    s = wmma_f16(a1.v, sb1.v, s);
    h8v ph;
#pragma unroll
    for (int r = 0; r < 8; ++r) {
      const float ca = cadj[t0 + ttl + r + hi * 8];
      ph[r] = (_Float16)__expf(s[r] + lbv - ca);
    }
    // C layout: lane->(M = r + hi*8, N = m16); store P' transposed: Pt[x][t], 8 consec t
    *(h8v*)&Pt[buf][xl + m16][ttl + hi * 8] = ph;
  };

  auto phaseB = [&](int t0, int buf) {
#pragma unroll
    for (int s2 = 0; s2 < 2; ++s2) {     // K-steps of 32 over the 64-t chunk
      U16 am[2], bm[2];
#pragma unroll
      for (int i = 0; i < 2; ++i) {      // c sub-tiles
        const _Float16* ap = Hh + (32 * w + 16 * i + m16) * T_DIM + t0 + 32 * s2 + hi * 16;
        am[i].h[0] = *(const h8v*)(ap);
        am[i].h[1] = *(const h8v*)(ap + 8);
      }
#pragma unroll
      for (int j = 0; j < 2; ++j) {      // x sub-tiles
        const _Float16* bp = &Pt[buf][16 * j + m16][32 * s2 + hi * 16];
        bm[j].h[0] = *(const h8v*)(bp);
        bm[j].h[1] = *(const h8v*)(bp + 8);
      }
#pragma unroll
      for (int i = 0; i < 2; ++i)
#pragma unroll
        for (int j = 0; j < 2; ++j)
          acc[i][j] = wmma_f16(am[i].v, bm[j].v, acc[i][j]);
    }
  };

  phaseA(0, 0);
  __syncthreads();
  for (int k = 0; k < T_DIM / 64; ++k) {
    const int t0 = k * 64;
    if (t0 + 64 < T_DIM) {
      // prefetch next Hh tiles into cache (global_prefetch_b8 path)
      __builtin_prefetch(Hh + (32 * w + m16) * T_DIM + t0 + 64, 0, 3);
      __builtin_prefetch(Hh + (32 * w + 16 + m16) * T_DIM + t0 + 64, 0, 3);
      phaseA(t0 + 64, (k + 1) & 1);      // writes buf (k+1)&1 (safe: last read 2 chunks ago)
    }
    phaseB(t0, k & 1);
    __syncthreads();
  }

  const float l2v = l2[0];
#pragma unroll
  for (int i = 0; i < 2; ++i)
#pragma unroll
    for (int j = 0; j < 2; ++j)
#pragma unroll
      for (int r = 0; r < 8; ++r) {
        const int c = 32 * w + 16 * i + r + hi * 8;
        const int x = xb + 16 * j + m16;
        Z[c * T_DIM + x] = l2v * acc[i][j][r];
      }
}

// ---------------------------------------------------------------------------
extern "C" void kernel_launch(void* const* d_in, const int* in_sizes, int n_in,
                              void* d_out, int out_size, void* d_ws, size_t ws_size,
                              hipStream_t stream) {
  (void)in_sizes; (void)n_in; (void)out_size; (void)ws_size;
  const float* H  = (const float*)d_in[0];   // [256, 8192]
  const float* D  = (const float*)d_in[1];   // [64, 256]
  const float* l2 = (const float*)d_in[2];   // scalar
  float* Z = (float*)d_out;                  // [256, 8192]

  char* ws = (char*)d_ws;                    // ~7.06 MB total
  _Float16* DHT  = (_Float16*)(ws + 0x000000);  // [8192][64] f16   (1 MiB)
  float*    DHf  = (float*)   (ws + 0x100000);  // [64][8192] f32   (2 MiB)
  float*    lbp  = (float*)   (ws + 0x300000);  // [8192] f32
  float*    cadj = (float*)   (ws + 0x308000);  // [8192] f32
  _Float16* Hh   = (_Float16*)(ws + 0x310000);  // [256][8192] f16  (4 MiB)

  k_dh      <<<dim3(T_DIM / 64, N_DIM / 4), dim3(64, 4), 0, stream>>>(D, H, DHf, DHT);
  k_lb      <<<T_DIM / 256, 256, 0, stream>>>(DHf, lbp);
  k_hh      <<<(D_DIM * T_DIM) / 256, 256, 0, stream>>>(H, Hh);
  k_rowstats<<<T_DIM / 128, 256, 0, stream>>>(DHT, lbp, cadj);
  k_attn    <<<T_DIM / 32, 256, 0, stream>>>(DHT, Hh, lbp, cadj, l2, Z);
}
